// SNNFabric_21741124453062
// MI455X (gfx1250) — matrix operand order repaired and checked by reading, not compiled, not run on hardware
//
#include <hip/hip_runtime.h>
#include <hip/hip_bf16.h>
#include <stdint.h>

typedef __attribute__((ext_vector_type(16))) _Float16 v16h;
typedef __attribute__((ext_vector_type(8)))  _Float16 v8h;
typedef __attribute__((ext_vector_type(8)))  float    v8f;

#define NB    32      // batch
#define NT    128     // timesteps
#define NE    8192
#define NI    2048
#define NR    32      // rank
#define NOUT  (NE + NI)
#define NSLAB 8
#define DECAY 0.90483741803595952f   // exp(-1/10)

union AFrag { v16h v; v8h h[2]; };

// ---- workspace layout (bytes, 256-aligned) ----
#define OFF_VE    ((size_t)0)                      // float [32][8192]
#define OFF_VI    (OFF_VE + (size_t)NB*NE*4)       // float [32][2048]
#define OFF_SE    (OFF_VI + (size_t)NB*NI*4)       // half  [32][8192]
#define OFF_SI    (OFF_SE + (size_t)NB*NE*2)       // half  [32][2048]
#define OFF_UPEE  (OFF_SI + (size_t)NB*NI*2)       // half  [8192][32]
#define OFF_UPIE  (OFF_UPEE + (size_t)NE*NR*2)     // half  [8192][32]
#define OFF_UPEI  (OFF_UPIE + (size_t)NE*NR*2)     // half  [2048][32]
#define OFF_VPEE  (OFF_UPEI + (size_t)NI*NR*2)     // half  [32][8192] (transposed V_ee)
#define OFF_VPEI  (OFF_VPEE + (size_t)NR*NE*2)     // half  [32][8192]
#define OFF_VPIE  (OFF_VPEI + (size_t)NR*NE*2)     // half  [32][2048]
#define OFF_PP    (OFF_VPIE + (size_t)NR*NI*2)     // float [3][8][32][32] slab partials
#define OFF_PF    (OFF_PP + (size_t)3*NSLAB*32*32*4) // half [3][32][32] reduced P
#define WS_NEEDED (OFF_PF + (size_t)3*32*32*2)

// Zero membrane state, spike state, and output accumulators (deterministic per call).
__global__ void k_init(float* vE, float* vI, _Float16* sE, _Float16* sI, float* out) {
    int i = blockIdx.x * blockDim.x + threadIdx.x;
    int stride = gridDim.x * blockDim.x;
    for (int j = i; j < NB * NE; j += stride) { vE[j] = 0.f; sE[j] = (_Float16)0.f; }
    for (int j = i; j < NB * NI; j += stride) { vI[j] = 0.f; sI[j] = (_Float16)0.f; }
    for (int j = i; j < NB * NOUT; j += stride) out[j] = 0.f;
}

// U stays row-major [N][R]; just convert to f16 (B-fragment reads are contiguous per lane).
__global__ void k_packU(const float* __restrict__ U, _Float16* __restrict__ Up, int n) {
    int i = blockIdx.x * blockDim.x + threadIdx.x;
    if (i < n) Up[i] = (_Float16)U[i];
}

// V is [K][R] row-major; transpose to Vp[r][k] f16 so each WMMA B lane loads 16
// contiguous halfs (one column r, 16 consecutive k).
__global__ void k_packVT(const float* __restrict__ V, _Float16* __restrict__ Vp, int K) {
    int k = blockIdx.x * blockDim.x + threadIdx.x;
    if (k >= K) return;
#pragma unroll
    for (int r = 0; r < NR; ++r)
        Vp[(size_t)r * K + k] = (_Float16)V[(size_t)k * NR + r];
}

// Phase 1: P[proj] = spikes @ V, K-slabbed. grid = (8 slabs, 3 projections), 4 waves.
// Each wave owns one 16x16 tile of the 32x32 output. K-loop unrolled 4x so 16
// global_load_b128 are in flight ahead of 4 chained WMMAs.
__global__ __launch_bounds__(128) void k_project(
    const _Float16* __restrict__ sE, const _Float16* __restrict__ sI,
    const _Float16* __restrict__ VpEE, const _Float16* __restrict__ VpEI,
    const _Float16* __restrict__ VpIE, float* __restrict__ Ppart) {
    const int proj = blockIdx.y;
    const int slab = blockIdx.x;
    const int wave = threadIdx.x >> 5;
    const int lane = threadIdx.x & 31;
    const int l = lane & 15, hi = lane >> 4;
    const int mtile = wave & 1, ntile = wave >> 1;

    const _Float16* A;  const _Float16* Bm;  int K;
    if (proj == 0)      { A = sE; Bm = VpEE; K = NE; }
    else if (proj == 1) { A = sE; Bm = VpEI; K = NE; }
    else                { A = sI; Bm = VpIE; K = NI; }

    const int slabK = K / NSLAB;              // 1024 or 256 (both multiples of 128)
    const int kb0 = slab * slabK;
    const _Float16* arow = A + (size_t)(mtile * 16 + l) * K;   // batch row m, f16 spikes
    const _Float16* brow = Bm + (size_t)(ntile * 16 + l) * K;  // rank column r (transposed)

    v8f acc = {};
    for (int kb = kb0; kb < kb0 + slabK; kb += 128) {
        AFrag a[4], b[4];
#pragma unroll
        for (int u = 0; u < 4; ++u) {
            const int k = kb + u * 32;
            // A 16x32 f16, k order used here (same as B): lane(hi=0) K 0..15, lane(hi=1) K 16..31 (contig per lane)
            a[u].h[0] = *(const v8h*)(arow + k + hi * 16);
            a[u].h[1] = *(const v8h*)(arow + k + hi * 16 + 8);
            // B 32x16 f16, k order used here (same as A): lanes0-15 K 0..15, lanes16-31 K 16..31 (contig per lane)
            b[u].v = *(const v16h*)(brow + k + hi * 16);
        }
#pragma unroll
        for (int u = 0; u < 4; ++u)
            acc = __builtin_amdgcn_wmma_f32_16x16x32_f16(false, a[u].v, false, b[u].v,
                                                         (short)0, acc, false, false);
    }
    // C/D f32 16x16: lane l holds N=l, vgpr v holds M = hi*8 + v
    float* dst = Ppart + (size_t)(proj * NSLAB + slab) * 32 * 32;
#pragma unroll
    for (int v = 0; v < 8; ++v)
        dst[(mtile * 16 + hi * 8 + v) * 32 + ntile * 16 + l] = acc[v];
}

// Reduce the 8 K-slab partials once per step into f16 [3][32][32] (A-fragment layout),
// so the 640 update blocks don't each redo ~64KB of reduction traffic.
__global__ void k_reduceP(const float* __restrict__ Ppart, _Float16* __restrict__ Pf16) {
    int e = blockIdx.x * blockDim.x + threadIdx.x;
    if (e >= 3 * 1024) return;
    int proj = e >> 10, i = e & 1023;
    float s = 0.f;
#pragma unroll
    for (int sl = 0; sl < NSLAB; ++sl)
        s += Ppart[(proj * NSLAB + sl) * 1024 + i];
    Pf16[e] = (_Float16)s;
}

// Phase 2: I = P @ U^T (+ ext in the C fragment), LIF update, spike emit, accumulate.
// P matrices are staged into LDS with gfx1250 async-to-LDS copies (ASYNCcnt-tracked),
// then read back as A-fragments via ds_load_b128.
__global__ __launch_bounds__(64) void k_update(
    const float* __restrict__ ext, const _Float16* __restrict__ Pf16,
    const _Float16* __restrict__ UpEE, const _Float16* __restrict__ UpIE,
    const _Float16* __restrict__ UpEI,
    float* __restrict__ vE, float* __restrict__ vI,
    _Float16* __restrict__ sE, _Float16* __restrict__ sI,
    float* __restrict__ out, int t) {
    __shared__ _Float16 P0s[32 * 32];
    __shared__ _Float16 P1s[32 * 32];

    const int tile = blockIdx.x;
    const bool isE = tile < (NE / 16);
    const int tid = threadIdx.x;

    // ---- async-stage P into LDS (16B per lane per op) ----
    {
        const _Float16* g0 = Pf16 + (isE ? 0 : 1) * 1024;
#pragma unroll
        for (int c = 0; c < 2; ++c) {
            unsigned dst = (unsigned)(uintptr_t)(void*)(P0s + c * 512 + tid * 8);
            unsigned long long src =
                (unsigned long long)(uintptr_t)(g0 + c * 512 + tid * 8);
            asm volatile("global_load_async_to_lds_b128 %0, %1, off"
                         :: "v"(dst), "v"(src) : "memory");
        }
        if (isE) {
            const _Float16* g1 = Pf16 + 2 * 1024;   // P_ie
#pragma unroll
            for (int c = 0; c < 2; ++c) {
                unsigned dst = (unsigned)(uintptr_t)(void*)(P1s + c * 512 + tid * 8);
                unsigned long long src =
                    (unsigned long long)(uintptr_t)(g1 + c * 512 + tid * 8);
                asm volatile("global_load_async_to_lds_b128 %0, %1, off"
                             :: "v"(dst), "v"(src) : "memory");
            }
        }
        asm volatile("s_wait_asynccnt 0x0" ::: "memory");
    }
    __syncthreads();

    const int mtile = tid >> 5;
    const int lane = tid & 31;
    const int l = lane & 15, hi = lane >> 4;

    AFrag a0;   // A fragment of P0 (rows mtile*16..+15, K = rank 0..31)
    {
        const _Float16* prow = P0s + (mtile * 16 + l) * 32;
        a0.h[0] = *(const v8h*)(prow + hi * 16);        // ds_load_b128
        a0.h[1] = *(const v8h*)(prow + hi * 16 + 8);
    }

    if (isE) {
        AFrag a1;
        {
            const _Float16* prow = P1s + (mtile * 16 + l) * 32;
            a1.h[0] = *(const v8h*)(prow + hi * 16);
            a1.h[1] = *(const v8h*)(prow + hi * 16 + 8);
        }
        const int n = tile * 16 + l;
        AFrag bee, bie;  // B = U^T: lane l -> neuron column n, 16 contiguous ranks
        bee.v = *(const v16h*)(UpEE + (size_t)n * NR + hi * 16);
        bie.v = *(const v16h*)(UpIE + (size_t)n * NR + hi * 16);

        // ext input goes straight into the C accumulator fragment
        const float* ep = ext + (size_t)t * NE + n;
        v8f c;
#pragma unroll
        for (int v = 0; v < 8; ++v)
            c[v] = ep[(size_t)(mtile * 16 + hi * 8 + v) * ((size_t)NT * NE)];
        if (t + 1 < NT) __builtin_prefetch(ep + NE, 0, 1);  // next step's ext row

        v8f d = __builtin_amdgcn_wmma_f32_16x16x32_f16(false, a1.v, false, bie.v,
                                                       (short)0, c, false, false);
        d = __builtin_amdgcn_wmma_f32_16x16x32_f16(false, a0.v, false, bee.v,
                                                   (short)0, d, false, false);
#pragma unroll
        for (int v = 0; v < 8; ++v) {
            int m = mtile * 16 + hi * 8 + v;
            size_t idx = (size_t)m * NE + n;
            float vm = vE[idx] * DECAY + d[v];
            float sp = (vm >= 1.0f) ? 1.0f : 0.0f;
            vE[idx] = vm * (1.0f - sp);
            sE[idx] = (_Float16)sp;
            out[(size_t)m * NOUT + n] += sp;
        }
    } else {
        const int n = (tile - NE / 16) * 16 + l;
        AFrag bei;
        bei.v = *(const v16h*)(UpEI + (size_t)n * NR + hi * 16);
        v8f c = {};
        v8f d = __builtin_amdgcn_wmma_f32_16x16x32_f16(false, a0.v, false, bei.v,
                                                       (short)0, c, false, false);
#pragma unroll
        for (int v = 0; v < 8; ++v) {
            int m = mtile * 16 + hi * 8 + v;
            size_t idx = (size_t)m * NI + n;
            float vm = vI[idx] * DECAY + d[v];
            float sp = (vm >= 1.0f) ? 1.0f : 0.0f;
            vI[idx] = vm * (1.0f - sp);
            sI[idx] = (_Float16)sp;
            out[(size_t)m * NOUT + NE + n] += sp;
        }
    }
}

extern "C" void kernel_launch(void* const* d_in, const int* in_sizes, int n_in,
                              void* d_out, int out_size, void* d_ws, size_t ws_size,
                              hipStream_t stream) {
    if (ws_size < WS_NEEDED) return;
    const float* ext = (const float*)d_in[0];
    const float* Uee = (const float*)d_in[1];
    const float* Vee = (const float*)d_in[2];
    const float* Uei = (const float*)d_in[3];
    const float* Vei = (const float*)d_in[4];
    const float* Uie = (const float*)d_in[5];
    const float* Vie = (const float*)d_in[6];

    char* ws = (char*)d_ws;
    float*    vE    = (float*)(ws + OFF_VE);
    float*    vI    = (float*)(ws + OFF_VI);
    _Float16* sE    = (_Float16*)(ws + OFF_SE);
    _Float16* sI    = (_Float16*)(ws + OFF_SI);
    _Float16* UpEE  = (_Float16*)(ws + OFF_UPEE);
    _Float16* UpIE  = (_Float16*)(ws + OFF_UPIE);
    _Float16* UpEI  = (_Float16*)(ws + OFF_UPEI);
    _Float16* VpEE  = (_Float16*)(ws + OFF_VPEE);
    _Float16* VpEI  = (_Float16*)(ws + OFF_VPEI);
    _Float16* VpIE  = (_Float16*)(ws + OFF_VPIE);
    float*    Ppart = (float*)(ws + OFF_PP);
    _Float16* Pf16  = (_Float16*)(ws + OFF_PF);
    float*    out   = (float*)d_out;

    k_init<<<256, 256, 0, stream>>>(vE, vI, sE, sI, out);
    k_packU<<<(NE * NR + 255) / 256, 256, 0, stream>>>(Uee, UpEE, NE * NR);
    k_packU<<<(NE * NR + 255) / 256, 256, 0, stream>>>(Uie, UpIE, NE * NR);
    k_packU<<<(NI * NR + 255) / 256, 256, 0, stream>>>(Uei, UpEI, NI * NR);
    k_packVT<<<(NE + 127) / 128, 128, 0, stream>>>(Vee, VpEE, NE);
    k_packVT<<<(NE + 127) / 128, 128, 0, stream>>>(Vei, VpEI, NE);
    k_packVT<<<(NI + 127) / 128, 128, 0, stream>>>(Vie, VpIE, NI);

    for (int t = 0; t < NT; ++t) {
        k_project<<<dim3(NSLAB, 3), 128, 0, stream>>>(sE, sI, VpEE, VpEI, VpIE, Ppart);
        k_reduceP<<<12, 256, 0, stream>>>(Ppart, Pf16);
        k_update<<<(NE + NI) / 16, 64, 0, stream>>>(ext, Pf16, UpEE, UpIE, UpEI,
                                                    vE, vI, sE, sI, out, t);
    }
}
